// MiniMaxM2DecoderLayer_8916352106891
// MI455X (gfx1250) — matrix operand order repaired and checked by reading, not compiled, not run on hardware
//
#include <hip/hip_runtime.h>
#include <cstdint>
#include <cstddef>

// ---------------- constants (B=1) ----------------
#define S_TOK 2048
#define DMODEL 2048
#define NH 16
#define NKV 4
#define HD_ 128
#define RD_ 64
#define NE 8
#define FF 1024
#define EF 8192            // NE * FF
#define QKV_N 3072         // NH*HD + 2*NKV*HD
#define KVD 512            // NKV*HD

typedef __attribute__((ext_vector_type(16))) __bf16 v16bf;
typedef __attribute__((ext_vector_type(8)))  float  v8f;
typedef __attribute__((ext_vector_type(4)))  unsigned int v4u;
typedef __attribute__((ext_vector_type(8)))  int    v8i;
typedef __attribute__((ext_vector_type(4)))  int    v4i;

union ABFrag { v16bf v; unsigned short u[16]; };
union CFrag  { v8f v; float f[8]; };

__device__ __forceinline__ unsigned short f32_to_bf16(float f) {
  unsigned int u = __float_as_uint(f);
  u += 0x7FFFu + ((u >> 16) & 1u);          // round-to-nearest-even
  return (unsigned short)(u >> 16);
}
__device__ __forceinline__ float bf16_to_f32(unsigned short h) {
  return __uint_as_float(((unsigned int)h) << 16);
}
__device__ __forceinline__ v8f vzero8() {
  v8f z = {0.f,0.f,0.f,0.f,0.f,0.f,0.f,0.f};
  return z;
}
__device__ __forceinline__ v8f wmma_bf16(v16bf a, v16bf b, v8f c) {
  // D = A(16x32 bf16) * B(32x16 bf16) + C(16x16 f32)   -> v_wmma_f32_16x16x32_bf16
  return __builtin_amdgcn_wmma_f32_16x16x32_bf16(false, a, false, b, (short)0, c, false, false);
}
// 16-bit A-matrix 16x32 VGPR layout (ISA 7.12.2): element e of per-lane v16bf
__device__ __forceinline__ int a_kidx(int e, int hi) {
  int p = e >> 1, w = e & 1;
  return ((p >= 4) ? 16 : 0) + (hi ? 8 : 0) + 2 * (p & 3) + w;
}
// 16-bit B-matrix 32x16 layout: same K order as A (lanes 0-15 K=0..7,16..23; lanes 16-31 K=8..15,24..31)
__device__ __forceinline__ int b_kidx(int e, int hi) {
  int p = e >> 1, w = e & 1;
  return ((p >= 4) ? 16 : 0) + (hi ? 8 : 0) + 2 * (p & 3) + w;
}

// ---------------- Tensor Data Mover: 2-D bf16 tile -> LDS (ISA ch.8 D#) ----------------
#if __has_builtin(__builtin_amdgcn_tensor_load_to_lds)
#define USE_TDM 1
#else
#define USE_TDM 0
#endif

#if USE_TDM
// Loads a tile_d1 x tile_d0 tile of 2-byte elements starting at gptr (row-major,
// row stride = stride0 elements) into LDS at lds_addr. When pad!=0: +1 dword
// after every 16 dwords (i.e. every 32-element row) => LDS row stride of 34 u16.
__device__ __forceinline__ void tdm_load_tile_2d(unsigned lds_addr, const void* gptr,
                                                 unsigned tensor_d0, unsigned tensor_d1,
                                                 unsigned stride0,
                                                 unsigned tile_d0, unsigned tile_d1,
                                                 int pad) {
  unsigned long long ga = (unsigned long long)(uintptr_t)gptr;
  v4u g0;
  g0[0] = 1u;                                        // count=1 valid descriptor
  g0[1] = lds_addr;                                  // lds_addr [63:32]
  g0[2] = (unsigned)(ga & 0xFFFFFFFFu);              // global_addr [95:64]
  g0[3] = (unsigned)((ga >> 32) & 0x01FFFFFFu)       // global_addr [120:96]
        | (2u << 30);                                // type=2 ("image") [127:126]
  v8i g1;
  unsigned w0 = (1u << 16);                          // data_size = 1 -> 2 bytes
  if (pad) w0 |= (1u << 20) | (3u << 22);            // pad_enable, interval=16 dwords
  g1[0] = (int)w0;
  g1[1] = (int)((tensor_d0 & 0xFFFFu) << 16);        // tensor_dim0[15:0] @ bits 63:48
  g1[2] = (int)(((tensor_d0 >> 16) & 0xFFFFu)        // tensor_dim0[31:16]
              | ((tensor_d1 & 0xFFFFu) << 16));      // tensor_dim1[15:0]
  g1[3] = (int)(((tensor_d1 >> 16) & 0xFFFFu)        // tensor_dim1[31:16]
              | ((tile_d0 & 0xFFFFu) << 16));        // tile_dim0 @ bits 127:112
  g1[4] = (int)(tile_d1 & 0xFFFFu);                  // tile_dim1 (tile_dim2 = 0)
  g1[5] = (int)stride0;                              // tensor_dim0_stride[31:0]
  g1[6] = 0;                                         // stride0[47:32], stride1[15:0]
  g1[7] = 0;
  v4i z4 = {0, 0, 0, 0};
#if defined(__clang_major__) && __clang_major__ >= 23
  v8i z8 = {0, 0, 0, 0, 0, 0, 0, 0};
  __builtin_amdgcn_tensor_load_to_lds(g0, g1, z4, z4, z8, 0);
#else
  __builtin_amdgcn_tensor_load_to_lds(g0, g1, z4, z4, 0);
#endif
}
#endif

// ---------------- elementwise / conversion kernels ----------------
__global__ void cvt_bf16_kernel(const float* __restrict__ in,
                                unsigned short* __restrict__ out, int n) {
  int i = blockIdx.x * 256 + threadIdx.x;
  if (i < n) out[i] = f32_to_bf16(in[i]);
}

// out[d*EF + e*FF + f] = w2[e, d, f]   (w2 is (E, D, F) row-major)
__global__ void cvt_w2t_kernel(const float* __restrict__ w2,
                               unsigned short* __restrict__ out) {
  int i = blockIdx.x * 256 + threadIdx.x;
  if (i >= DMODEL * EF) return;
  int d = i >> 13;           // / EF
  int ef = i & (EF - 1);
  int e = ef >> 10;          // / FF
  int f = ef & (FF - 1);
  out[i] = f32_to_bf16(w2[((size_t)e * DMODEL + d) * FF + f]);
}

// resid_out (optional) = x + add ; out_bf16 = rmsnorm(x+add) * w
__launch_bounds__(256)
__global__ void add_rmsnorm_kernel(const float* __restrict__ x,
                                   const float* __restrict__ addend,
                                   const float* __restrict__ w,
                                   float* __restrict__ resid_out,
                                   unsigned short* __restrict__ out_bf16) {
  __shared__ float red[256];
  const int row = blockIdx.x;
  const int tid = threadIdx.x;
  const float* xr = x + (size_t)row * DMODEL;
  const float* ar = addend ? addend + (size_t)row * DMODEL : nullptr;
  float ss = 0.f;
  for (int j = tid; j < DMODEL; j += 256) {
    float v = xr[j] + (ar ? ar[j] : 0.f);
    ss += v * v;
  }
  red[tid] = ss; __syncthreads();
  for (int s = 128; s > 0; s >>= 1) {
    if (tid < s) red[tid] += red[tid + s];
    __syncthreads();
  }
  const float r = rsqrtf(red[0] * (1.0f / DMODEL) + 1e-6f);
  for (int j = tid; j < DMODEL; j += 256) {
    float v = xr[j] + (ar ? ar[j] : 0.f);
    if (resid_out) resid_out[(size_t)row * DMODEL + j] = v;
    out_bf16[(size_t)row * DMODEL + j] = f32_to_bf16(v * w[j]);
  }
}

// per-token: full-row rmsnorm of q (2048) and k (512), partial RoPE, split into
// head-major bf16 buffers Q(H,S,HD) K(HKV,S,HD) V(HKV,S,HD)
__launch_bounds__(128)
__global__ void qk_norm_rope_kernel(const float* __restrict__ qkv,
                                    const float* __restrict__ cosp,
                                    const float* __restrict__ sinp,
                                    const float* __restrict__ qn_w,
                                    const float* __restrict__ kn_w,
                                    unsigned short* __restrict__ Qo,
                                    unsigned short* __restrict__ Ko,
                                    unsigned short* __restrict__ Vo) {
  __shared__ float red[128];
  __shared__ float rq_s, rk_s;
  const int t = blockIdx.x;
  const int tid = threadIdx.x;
  const float* row = qkv + (size_t)t * QKV_N;

  float sq = 0.f;
  for (int j = tid; j < DMODEL; j += 128) sq += row[j] * row[j];
  red[tid] = sq; __syncthreads();
  for (int s = 64; s > 0; s >>= 1) { if (tid < s) red[tid] += red[tid + s]; __syncthreads(); }
  if (tid == 0) rq_s = rsqrtf(red[0] * (1.0f / DMODEL) + 1e-6f);
  __syncthreads();
  float sk = 0.f;
  for (int j = tid; j < KVD; j += 128) { float v = row[DMODEL + j]; sk += v * v; }
  red[tid] = sk; __syncthreads();
  for (int s = 64; s > 0; s >>= 1) { if (tid < s) red[tid] += red[tid + s]; __syncthreads(); }
  if (tid == 0) rk_s = rsqrtf(red[0] * (1.0f / KVD) + 1e-6f);
  __syncthreads();
  const float rq = rq_s, rk = rk_s;

  for (int j = tid; j < DMODEL; j += 128) {   // Q
    int h = j >> 7, d = j & 127;
    float v = row[j] * rq * qn_w[j];
    float o = v;
    if (d < RD_) {
      int pj = (d < 32) ? j + 32 : j - 32;
      float pv = row[pj] * rq * qn_w[pj];
      float rot = (d < 32) ? -pv : pv;
      o = v * cosp[t * RD_ + d] + rot * sinp[t * RD_ + d];
    }
    Qo[((size_t)h * S_TOK + t) * HD_ + d] = f32_to_bf16(o);
  }
  for (int j = tid; j < KVD; j += 128) {      // K
    int h = j >> 7, d = j & 127;
    float v = row[DMODEL + j] * rk * kn_w[j];
    float o = v;
    if (d < RD_) {
      int pj = (d < 32) ? j + 32 : j - 32;
      float pv = row[DMODEL + pj] * rk * kn_w[pj];
      float rot = (d < 32) ? -pv : pv;
      o = v * cosp[t * RD_ + d] + rot * sinp[t * RD_ + d];
    }
    Ko[((size_t)h * S_TOK + t) * HD_ + d] = f32_to_bf16(o);
  }
  for (int j = tid; j < KVD; j += 128) {      // V
    int h = j >> 7, d = j & 127;
    Vo[((size_t)h * S_TOK + t) * HD_ + d] = f32_to_bf16(row[DMODEL + KVD + j]);
  }
}

// ---------------- tiled bf16 WMMA GEMM:  C(MxN,f32) = A(MxK) * W(NxK)^T ----------------
// Double-buffered TDM pipeline: wave 0 issues the NEXT k-tile's two
// TENSOR_LOAD_TO_LDS descriptors into the alternate LDS buffer, then waits
// s_wait_tensorcnt 2 (current tile resident) so DMA overlaps the WMMA work.
#define GBM 128
#define GBN 64
#define GBK 32
__launch_bounds__(256)
__global__ void gemm_bf16_nt(const unsigned short* __restrict__ A,
                             const unsigned short* __restrict__ W,
                             float* __restrict__ C,
                             int M, int N, int K) {
  __shared__ unsigned short As[2][GBM][GBK + 2];   // row stride 34 u16 = 17 dwords
  __shared__ unsigned short Bs[2][GBN][GBK + 2];
  const int tid  = threadIdx.x;
  const int wave = tid >> 5;
  const int lane = tid & 31;
  const int ln16 = lane & 15;
  const int hi   = lane >> 4;
  const int rowBase = blockIdx.y * GBM;
  const int colBase = blockIdx.x * GBN;

  v8f acc[4];
  #pragma unroll
  for (int i = 0; i < 4; ++i) acc[i] = vzero8();

  const int ntile = K / GBK;
#if USE_TDM
  if (wave == 0) {   // prologue: tile 0 into buffer 0
    tdm_load_tile_2d((unsigned)(uintptr_t)&As[0][0][0], A + (size_t)rowBase * K,
                     (unsigned)K, (unsigned)M, (unsigned)K, GBK, GBM, 1);
    tdm_load_tile_2d((unsigned)(uintptr_t)&Bs[0][0][0], W + (size_t)colBase * K,
                     (unsigned)K, (unsigned)N, (unsigned)K, GBK, GBN, 1);
  }
  int cur = 0;
  for (int it = 0; it < ntile; ++it) {
    if (wave == 0) {
      if (it + 1 < ntile) {   // issue next tile into the other buffer
        const int kb2 = (it + 1) * GBK;
        tdm_load_tile_2d((unsigned)(uintptr_t)&As[cur ^ 1][0][0],
                         A + (size_t)rowBase * K + kb2,
                         (unsigned)K, (unsigned)M, (unsigned)K, GBK, GBM, 1);
        tdm_load_tile_2d((unsigned)(uintptr_t)&Bs[cur ^ 1][0][0],
                         W + (size_t)colBase * K + kb2,
                         (unsigned)K, (unsigned)N, (unsigned)K, GBK, GBN, 1);
        __builtin_amdgcn_s_wait_tensorcnt(2);   // current tile's 2 loads done
      } else {
        __builtin_amdgcn_s_wait_tensorcnt(0);
      }
    }
    __syncthreads();

    ABFrag af;
    #pragma unroll
    for (int e = 0; e < 16; ++e)
      af.u[e] = As[cur][wave * 16 + ln16][a_kidx(e, hi)];
    #pragma unroll
    for (int nt = 0; nt < 4; ++nt) {
      ABFrag bf;
      #pragma unroll
      for (int e = 0; e < 16; ++e)
        bf.u[e] = Bs[cur][nt * 16 + ln16][b_kidx(e, hi)];
      acc[nt] = wmma_bf16(af.v, bf.v, acc[nt]);
    }
    __syncthreads();
    cur ^= 1;
  }
#else
  for (int it = 0; it < ntile; ++it) {
    const int kb = it * GBK;
    if (kb + GBK < K)
      __builtin_prefetch(A + (size_t)(rowBase + (tid >> 1)) * K + kb + GBK, 0, 1);
    #pragma unroll
    for (int i = 0; i < 8; ++i) {           // A tile: 128x32, pairs of u16
      int p = tid + i * 256;
      int r = p >> 4;
      int cp = (p & 15) << 1;
      const unsigned short* src = A + (size_t)(rowBase + r) * K + kb + cp;
      As[0][r][cp]     = src[0];
      As[0][r][cp + 1] = src[1];
    }
    #pragma unroll
    for (int i = 0; i < 4; ++i) {           // B tile: 64x32
      int p = tid + i * 256;
      int r = p >> 4;
      int cp = (p & 15) << 1;
      const unsigned short* src = W + (size_t)(colBase + r) * K + kb + cp;
      Bs[0][r][cp]     = src[0];
      Bs[0][r][cp + 1] = src[1];
    }
    __syncthreads();
    ABFrag af;
    #pragma unroll
    for (int e = 0; e < 16; ++e)
      af.u[e] = As[0][wave * 16 + ln16][a_kidx(e, hi)];
    #pragma unroll
    for (int nt = 0; nt < 4; ++nt) {
      ABFrag bf;
      #pragma unroll
      for (int e = 0; e < 16; ++e)
        bf.u[e] = Bs[0][nt * 16 + ln16][b_kidx(e, hi)];
      acc[nt] = wmma_bf16(af.v, bf.v, acc[nt]);
    }
    __syncthreads();
  }
#endif

  #pragma unroll
  for (int nt = 0; nt < 4; ++nt) {
    CFrag cf; cf.v = acc[nt];
    #pragma unroll
    for (int v = 0; v < 8; ++v) {
      int r = rowBase + wave * 16 + v + (hi ? 8 : 0);
      int c = colBase + nt * 16 + ln16;
      C[(size_t)r * N + c] = cf.f[v];
    }
  }
}

// ---------------- flash attention: one wave32 per (16-query tile, head) ----------------
// K/V 32x128 bf16 tiles are staged into LDS by the Tensor Data Mover each
// key-block; the wave waits s_wait_tensorcnt 0 then builds WMMA fragments
// from LDS instead of strided global loads.
__launch_bounds__(32)
__global__ void flash_attn_kernel(const unsigned short* __restrict__ Q,   // (H,S,HD)
                                  const unsigned short* __restrict__ Kc,  // (HKV,S,HD)
                                  const unsigned short* __restrict__ V,   // (HKV,S,HD)
                                  unsigned short* __restrict__ ctx) {     // (S, H*HD)
#if USE_TDM
  __shared__ unsigned short Kl[32][HD_];
  __shared__ unsigned short Vl[32][HD_];
#endif
  __shared__ unsigned short Pl[16][34];
  const int lane  = threadIdx.x;
  const int ln16  = lane & 15;
  const int hi    = lane >> 4;
  const int qbase = blockIdx.x * 16;
  const int h     = blockIdx.y;
  const int kv    = h >> 2;                 // H / HKV = 4
  const float inv_sqrt_d = 0.08838834764831845f;
  const float NEGINF = -__builtin_inff();

  ABFrag qf[4];
  const unsigned short* qrow = Q + ((size_t)h * S_TOK + qbase + ln16) * HD_;
  #pragma unroll
  for (int c = 0; c < 4; ++c)
    #pragma unroll
    for (int e = 0; e < 16; ++e)
      qf[c].u[e] = qrow[c * 32 + a_kidx(e, hi)];

  v8f acc[8];
  #pragma unroll
  for (int c = 0; c < 8; ++c) acc[c] = vzero8();
  float m[8], l[8];
  #pragma unroll
  for (int v = 0; v < 8; ++v) { m[v] = NEGINF; l[v] = 0.f; }

  const int kEnd = qbase + 16;
  for (int kb = 0; kb < kEnd; kb += 32) {
    // S = Q * K^T over 32 keys (two 16-key column tiles)
    v8f s0 = vzero8(), s1 = vzero8();
#if USE_TDM
    tdm_load_tile_2d((unsigned)(uintptr_t)&Kl[0][0],
                     Kc + ((size_t)kv * S_TOK + kb) * HD_,
                     HD_, (unsigned)S_TOK, HD_, HD_, 32, 0);
    tdm_load_tile_2d((unsigned)(uintptr_t)&Vl[0][0],
                     V + ((size_t)kv * S_TOK + kb) * HD_,
                     HD_, (unsigned)S_TOK, HD_, HD_, 32, 0);
    __builtin_amdgcn_s_wait_tensorcnt(0);
    #pragma unroll
    for (int c = 0; c < 4; ++c) {
      ABFrag b0, b1;
      #pragma unroll
      for (int e = 0; e < 16; ++e) {
        int d = c * 32 + b_kidx(e, hi);
        b0.u[e] = Kl[ln16][d];
        b1.u[e] = Kl[16 + ln16][d];
      }
      s0 = wmma_bf16(qf[c].v, b0.v, s0);
      s1 = wmma_bf16(qf[c].v, b1.v, s1);
    }
#else
    if (kb + 32 < kEnd) {
      __builtin_prefetch(Kc + ((size_t)kv * S_TOK + kb + 32 + lane) * HD_, 0, 1);
      __builtin_prefetch(V  + ((size_t)kv * S_TOK + kb + 32 + lane) * HD_, 0, 1);
    }
    #pragma unroll
    for (int c = 0; c < 4; ++c) {
      ABFrag b0, b1;
      const unsigned short* kp0 = Kc + ((size_t)kv * S_TOK + kb + ln16) * HD_ + c * 32;
      const unsigned short* kp1 = kp0 + (size_t)16 * HD_;
      #pragma unroll
      for (int e = 0; e < 16; ++e) {
        int d = b_kidx(e, hi);
        b0.u[e] = kp0[d];
        b1.u[e] = kp1[d];
      }
      s0 = wmma_bf16(qf[c].v, b0.v, s0);
      s1 = wmma_bf16(qf[c].v, b1.v, s1);
    }
#endif

    // online softmax update
    CFrag c0, c1; c0.v = s0; c1.v = s1;
    const int key0 = kb + ln16;
    const int key1 = kb + 16 + ln16;
    float p0[8], p1[8], scv[8];
    #pragma unroll
    for (int v = 0; v < 8; ++v) {
      const int row = qbase + v + (hi ? 8 : 0);
      float a0 = (key0 <= row) ? c0.f[v] * inv_sqrt_d : NEGINF;
      float a1 = (key1 <= row) ? c1.f[v] * inv_sqrt_d : NEGINF;
      float mv = fmaxf(a0, a1);
      #pragma unroll
      for (int off = 1; off < 16; off <<= 1)
        mv = fmaxf(mv, __shfl_xor(mv, off, 16));
      const float mn = fmaxf(m[v], mv);
      scv[v] = __expf(m[v] - mn);
      p0[v] = __expf(a0 - mn);
      p1[v] = __expf(a1 - mn);
      float rs = p0[v] + p1[v];
      #pragma unroll
      for (int off = 1; off < 16; off <<= 1)
        rs += __shfl_xor(rs, off, 16);
      l[v] = l[v] * scv[v] + rs;
      m[v] = mn;
    }
    #pragma unroll
    for (int c = 0; c < 8; ++c) {
      CFrag t; t.v = acc[c];
      #pragma unroll
      for (int v = 0; v < 8; ++v) t.f[v] *= scv[v];
      acc[c] = t.v;
    }

    // relayout P (C-frag f32) -> LDS -> A-frag bf16
    __syncthreads();
    #pragma unroll
    for (int v = 0; v < 8; ++v) {
      Pl[v + hi * 8][ln16]      = f32_to_bf16(p0[v]);
      Pl[v + hi * 8][16 + ln16] = f32_to_bf16(p1[v]);
    }
    __syncthreads();
    ABFrag pf;
    #pragma unroll
    for (int e = 0; e < 16; ++e)
      pf.u[e] = Pl[ln16][a_kidx(e, hi)];

    // O += P * V
    #pragma unroll
    for (int c = 0; c < 8; ++c) {
      ABFrag vf;
      #pragma unroll
      for (int e = 0; e < 16; ++e) {
#if USE_TDM
        vf.u[e] = Vl[b_kidx(e, hi)][c * 16 + ln16];
#else
        vf.u[e] = V[((size_t)kv * S_TOK + kb + b_kidx(e, hi)) * HD_ + c * 16 + ln16];
#endif
      }
      acc[c] = wmma_bf16(pf.v, vf.v, acc[c]);
    }
  }

  // epilogue: normalize by l, write bf16 context (S, H*HD)
  #pragma unroll
  for (int c = 0; c < 8; ++c) {
    CFrag t; t.v = acc[c];
    #pragma unroll
    for (int v = 0; v < 8; ++v) {
      int row = qbase + v + hi * 8;
      int col = h * HD_ + c * 16 + ln16;
      ctx[(size_t)row * DMODEL + col] = f32_to_bf16(t.f[v] / l[v]);
    }
  }
}

// ---------------- MoE routing (sigmoid + top-2) ----------------
__launch_bounds__(256)
__global__ void moe_gate_kernel(const unsigned short* __restrict__ xn,  // bf16 (T,D)
                                const float* __restrict__ gate_w,       // (E,D)
                                const float* __restrict__ bias_corr,    // (E,)
                                float* __restrict__ comb) {             // (T,E)
  __shared__ float red[256];
  __shared__ float logit_s[NE];
  const int t = blockIdx.x;
  const int tid = threadIdx.x;
  for (int e = 0; e < NE; ++e) {
    float p = 0.f;
    for (int j = tid; j < DMODEL; j += 256)
      p += bf16_to_f32(xn[(size_t)t * DMODEL + j]) * gate_w[(size_t)e * DMODEL + j];
    red[tid] = p; __syncthreads();
    for (int s = 128; s > 0; s >>= 1) { if (tid < s) red[tid] += red[tid + s]; __syncthreads(); }
    if (tid == 0) logit_s[e] = red[0];
    __syncthreads();
  }
  if (tid == 0) {
    float rw[NE], sc[NE];
    for (int e = 0; e < NE; ++e) {
      rw[e] = 1.f / (1.f + __expf(-logit_s[e]));
      sc[e] = rw[e] + bias_corr[e];
    }
    int i0 = 0;
    for (int e = 1; e < NE; ++e) if (sc[e] > sc[i0]) i0 = e;
    int i1 = (i0 == 0) ? 1 : 0;
    for (int e = 0; e < NE; ++e) if (e != i0 && sc[e] > sc[i1]) i1 = e;
    const float w0 = rw[i0], w1 = rw[i1], s = w0 + w1;
    for (int e = 0; e < NE; ++e) comb[(size_t)t * NE + e] = 0.f;
    comb[(size_t)t * NE + i0] = w0 / s;
    comb[(size_t)t * NE + i1] = w1 / s;
  }
}

// h' = silu(gate) * up * comb[t, e]   -> bf16, (T, E*F)
__global__ void silu_combine_kernel(const float* __restrict__ g,
                                    const float* __restrict__ u,
                                    const float* __restrict__ comb,
                                    unsigned short* __restrict__ out, int n) {
  int i = blockIdx.x * 256 + threadIdx.x;
  if (i >= n) return;
  int t = i >> 13;            // / EF
  int e = (i >> 10) & 7;      // expert
  float gv = g[i];
  float hv = gv / (1.f + __expf(-gv)) * u[i] * comb[(size_t)t * NE + e];
  out[i] = f32_to_bf16(hv);
}

__global__ void add2_kernel(const float* __restrict__ a,
                            const float* __restrict__ b,
                            float* __restrict__ o, int n) {
  int i = blockIdx.x * 256 + threadIdx.x;
  if (i < n) o[i] = a[i] + b[i];
}

// ---------------- host orchestration ----------------
extern "C" void kernel_launch(void* const* d_in, const int* in_sizes, int n_in,
                              void* d_out, int out_size, void* d_ws, size_t ws_size,
                              hipStream_t stream) {
  (void)in_sizes; (void)n_in; (void)out_size; (void)ws_size;
  const float* x    = (const float*)d_in[0];
  const float* cosp = (const float*)d_in[1];
  const float* sinp = (const float*)d_in[2];
  const float* ln1  = (const float*)d_in[3];
  const float* ln2  = (const float*)d_in[4];
  const float* qw   = (const float*)d_in[5];
  const float* kw   = (const float*)d_in[6];
  const float* vw   = (const float*)d_in[7];
  const float* ow   = (const float*)d_in[8];
  const float* qnw  = (const float*)d_in[9];
  const float* knw  = (const float*)d_in[10];
  const float* gw   = (const float*)d_in[11];
  const float* w1   = (const float*)d_in[12];
  const float* w2   = (const float*)d_in[13];
  const float* w3   = (const float*)d_in[14];
  const float* bias = (const float*)d_in[15];
  float* out = (float*)d_out;

  char* ws = (char*)d_ws;
  size_t off = 0;
  auto alloc = [&](size_t bytes) -> char* {
    char* p = ws + off;
    off += (bytes + 255) & ~(size_t)255;
    return p;
  };

  unsigned short* xn1   = (unsigned short*)alloc((size_t)S_TOK * DMODEL * 2);
  unsigned short* wqkvb = (unsigned short*)alloc((size_t)QKV_N * DMODEL * 2);
  unsigned short* wob   = (unsigned short*)alloc((size_t)DMODEL * DMODEL * 2);
  unsigned short* w1b   = (unsigned short*)alloc((size_t)EF * DMODEL * 2);
  unsigned short* w3b   = (unsigned short*)alloc((size_t)EF * DMODEL * 2);
  unsigned short* w2tb  = (unsigned short*)alloc((size_t)DMODEL * EF * 2);
  float*          qkv   = (float*)alloc((size_t)S_TOK * QKV_N * 4);
  unsigned short* Qb    = (unsigned short*)alloc((size_t)NH * S_TOK * HD_ * 2);
  unsigned short* Kb    = (unsigned short*)alloc((size_t)NKV * S_TOK * HD_ * 2);
  unsigned short* Vb    = (unsigned short*)alloc((size_t)NKV * S_TOK * HD_ * 2);
  unsigned short* ctxb  = (unsigned short*)alloc((size_t)S_TOK * DMODEL * 2);
  float*          attn  = (float*)alloc((size_t)S_TOK * DMODEL * 4);
  float*          res1  = (float*)alloc((size_t)S_TOK * DMODEL * 4);
  unsigned short* xn2   = (unsigned short*)alloc((size_t)S_TOK * DMODEL * 2);
  float*          combp = (float*)alloc((size_t)S_TOK * NE * 4);
  float*          gbuf  = (float*)alloc((size_t)S_TOK * EF * 4);
  float*          ubuf  = (float*)alloc((size_t)S_TOK * EF * 4);
  unsigned short* hpb   = (unsigned short*)alloc((size_t)S_TOK * EF * 2);
  float*          moeo  = (float*)alloc((size_t)S_TOK * DMODEL * 4);

  auto blocks = [](int n) { return (n + 255) / 256; };

  // --- weight conversions (bf16) ---
  cvt_bf16_kernel<<<blocks(DMODEL * DMODEL), 256, 0, stream>>>(qw, wqkvb, DMODEL * DMODEL);
  cvt_bf16_kernel<<<blocks(KVD * DMODEL), 256, 0, stream>>>(kw, wqkvb + (size_t)DMODEL * DMODEL, KVD * DMODEL);
  cvt_bf16_kernel<<<blocks(KVD * DMODEL), 256, 0, stream>>>(vw, wqkvb + (size_t)(DMODEL + KVD) * DMODEL, KVD * DMODEL);
  cvt_bf16_kernel<<<blocks(DMODEL * DMODEL), 256, 0, stream>>>(ow, wob, DMODEL * DMODEL);
  cvt_bf16_kernel<<<blocks(EF * DMODEL), 256, 0, stream>>>(w1, w1b, EF * DMODEL);
  cvt_bf16_kernel<<<blocks(EF * DMODEL), 256, 0, stream>>>(w3, w3b, EF * DMODEL);
  cvt_w2t_kernel<<<blocks(DMODEL * EF), 256, 0, stream>>>(w2, w2tb);

  // --- attention path ---
  add_rmsnorm_kernel<<<S_TOK, 256, 0, stream>>>(x, nullptr, ln1, nullptr, xn1);
  gemm_bf16_nt<<<dim3(QKV_N / GBN, S_TOK / GBM), 256, 0, stream>>>(xn1, wqkvb, qkv, S_TOK, QKV_N, DMODEL);
  qk_norm_rope_kernel<<<S_TOK, 128, 0, stream>>>(qkv, cosp, sinp, qnw, knw, Qb, Kb, Vb);
  flash_attn_kernel<<<dim3(S_TOK / 16, NH), 32, 0, stream>>>(Qb, Kb, Vb, ctxb);
  gemm_bf16_nt<<<dim3(DMODEL / GBN, S_TOK / GBM), 256, 0, stream>>>(ctxb, wob, attn, S_TOK, DMODEL, DMODEL);

  // --- residual + second norm ---
  add_rmsnorm_kernel<<<S_TOK, 256, 0, stream>>>(x, attn, ln2, res1, xn2);

  // --- MoE ---
  moe_gate_kernel<<<S_TOK, 256, 0, stream>>>(xn2, gw, bias, combp);
  gemm_bf16_nt<<<dim3(EF / GBN, S_TOK / GBM), 256, 0, stream>>>(xn2, w1b, gbuf, S_TOK, EF, DMODEL);
  gemm_bf16_nt<<<dim3(EF / GBN, S_TOK / GBM), 256, 0, stream>>>(xn2, w3b, ubuf, S_TOK, EF, DMODEL);
  silu_combine_kernel<<<blocks(S_TOK * EF), 256, 0, stream>>>(gbuf, ubuf, combp, hpb, S_TOK * EF);
  gemm_bf16_nt<<<dim3(DMODEL / GBN, S_TOK / GBM), 256, 0, stream>>>(hpb, w2tb, moeo, S_TOK, DMODEL, EF);

  // --- final residual ---
  add2_kernel<<<blocks(S_TOK * DMODEL), 256, 0, stream>>>(res1, moeo, out, S_TOK * DMODEL);
}
